// NeuralODE_15951508537805
// MI455X (gfx1250) — matrix operand order repaired and checked by reading, not compiled, not run on hardware
//
#include <hip/hip_runtime.h>
#include <hip/hip_fp16.h>
#include <math.h>

typedef __attribute__((ext_vector_type(16))) _Float16 v16h;
typedef __attribute__((ext_vector_type(8)))  _Float16 v8h;
typedef __attribute__((ext_vector_type(8)))  float    v8f;

namespace {
constexpr int kBatch   = 4096;
constexpr int kD       = 64;    // state dim
constexpr int kH       = 256;   // hidden dim
constexpr int kT       = 128;   // save points
constexpr int kBM      = 32;    // batch rows per workgroup
constexpr int kThreads = 256;   // 8 wave32

// LDS row pitches in halves (+8 pad -> 144B / 528B row stride, conflict-free b128 frag loads)
constexpr int LDA  = kD + 8;    // stage-input f16 [kBM][kD]
constexpr int LDH  = kH + 8;    // hidden f16 [kBM][kH]
constexpr int LDW1 = kD + 8;    // W1^T f16 [kH][kD]
constexpr int LDW2 = kH + 8;    // W2^T f16 [kD][kH]

constexpr size_t kSmemHalves = (size_t)kH * LDW1 + (size_t)kD * LDW2 +
                               (size_t)kBM * LDA + (size_t)kBM * LDH;
constexpr size_t kSmemBytes  = kSmemHalves * sizeof(_Float16) + (kH + kD) * sizeof(float);

// Tsit5 tableau
constexpr float cA21 = 0.161f;
constexpr float cA31 = -0.008480655492356989f, cA32 = 0.335480655492357f;
constexpr float cA41 = 2.8971530571054935f,  cA42 = -6.359448489975075f,  cA43 = 4.3622954328695815f;
constexpr float cA51 = 5.325864828439257f,   cA52 = -11.748883564062828f, cA53 = 7.4955393428898365f, cA54 = -0.09249506636175525f;
constexpr float cA61 = 5.86145544294642f,    cA62 = -12.92096931784711f,  cA63 = 8.159367898576159f,  cA64 = -0.071584973281401f, cA65 = -0.028269050394068383f;
constexpr float cB1 = 0.09646076681806523f, cB2 = 0.01f, cB3 = 0.4798896504144996f;
constexpr float cB4 = 1.379008574103742f,   cB5 = -3.290069515436081f, cB6 = 2.324710524099774f;
} // namespace

// A-fragment (16x32 f16, ISA 7.12.2): lane l holds row l&15; lanes 0-15 get K=k0+0..7
// in VGPR0-3 and K=k0+16..23 in VGPR4-7; lanes 16-31 shifted by +8. Two ds_load_b128.
__device__ __forceinline__ v16h load_fragA(const _Float16* __restrict__ base, int ld,
                                           int row0, int k0, int lane) {
  const int r  = lane & 15;
  const int kg = (lane >> 4) << 3;           // 0 or 8
  const _Float16* p = base + (row0 + r) * ld + k0 + kg;
  union { v16h v; v8h h[2]; } u;
  u.h[0] = *(const v8h*)(p);                 // K = k0+kg   .. +7
  u.h[1] = *(const v8h*)(p + 16);            // K = k0+16+kg.. +7
  return u.v;
}

// B-fragment (32x16 f16): lane l holds column l&15; lanes 0-15 K=k0+0..7 and
// K=k0+16..23, lanes 16-31 shifted by +8 (same as A-fragment). Two ds_load_b128.
__device__ __forceinline__ v16h load_fragB(const _Float16* __restrict__ base, int ld,
                                           int col0, int k0, int lane) {
  const int c  = lane & 15;
  const int kg = (lane >> 4) << 3;           // 0 or 8
  const _Float16* p = base + (col0 + c) * ld + k0 + kg;
  union { v16h v; v8h h[2]; } u;
  u.h[0] = *(const v8h*)(p);
  u.h[1] = *(const v8h*)(p + 16);
  return u.v;
}

__device__ __forceinline__ v8f wmma_f16(v16h a, v16h b, v8f c) {
  // D = A*B + C, f32 accumulate -> v_wmma_f32_16x16x32_f16
  return __builtin_amdgcn_wmma_f32_16x16x32_f16(false, a, false, b, (short)0, c, false, false);
}

// Packed-f16 tanh for a pair of fp32 WMMA accumulators. The result feeds the next
// f16 WMMA anyway, so f16 internal precision is sufficient; packed ops (v_pk_*_f16)
// halve the VALU count on the serial chain and the TRANS ops become v_exp_f16/v_rcp_f16.
// tanh(x) = sign(x) * (1-e)/(1+e),  e = exp2(-2*log2(e)*|x|).
__device__ __forceinline__ __half2 tanh2(float a, float b) {
  __half2 x   = __floats2half2_rn(a, b);
  __half2 ax  = __habs2(x);
  const __half2 kc  = __float2half2_rn(-2.8853900817779268f);  // -2/ln(2)
  const __half2 one = __float2half2_rn(1.0f);
  __half2 e   = h2exp2(__hmul2(ax, kc));
  __half2 num = __hsub2(one, e);
  __half2 den = __hadd2(one, e);
  __half2 r   = __hmul2(num, h2rcp(den));
  // copysign on both halves in one v_bfi_b32
  unsigned xb, rb;
  __builtin_memcpy(&xb, &x, 4);
  __builtin_memcpy(&rb, &r, 4);
  unsigned res = (rb & 0x7fff7fffu) | (xb & 0x80008000u);
  __half2 out;
  __builtin_memcpy(&out, &res, 4);
  return out;
}

__global__ void __launch_bounds__(kThreads)
node_tsit5_kernel(const float* __restrict__ y0,  const float* __restrict__ ts,
                  const float* __restrict__ W1,  const float* __restrict__ b1,
                  const float* __restrict__ W2,  const float* __restrict__ b2,
                  const int*   __restrict__ nsubp, float* __restrict__ out) {
  extern __shared__ char smem_raw[];
  _Float16* sW1t = (_Float16*)smem_raw;            // [kH][LDW1]  W1 transposed
  _Float16* sW2t = sW1t + kH * LDW1;               // [kD][LDW2]  W2 transposed
  _Float16* sA   = sW2t + kD * LDW2;               // [kBM][LDA]  stage input (f16)
  _Float16* sAh  = sA   + kBM * LDA;               // [kBM][LDH]  hidden activations (f16)
  float*    sb1  = (float*)(sAh + kBM * LDH);
  float*    sb2  = sb1 + kH;

  const int tid = threadIdx.x;
  const int wg  = blockIdx.x;

  // One-time: stage f16-transposed weights + biases into LDS.
  for (int i = tid; i < kD * kH; i += kThreads) {  // W1[k][n] -> W1t[n][k]
    int k = i / kH, n = i % kH;
    sW1t[n * LDW1 + k] = (_Float16)W1[i];
  }
  for (int i = tid; i < kH * kD; i += kThreads) {  // W2[k][n] -> W2t[n][k]
    int k = i / kD, n = i % kD;
    sW2t[n * LDW2 + k] = (_Float16)W2[i];
  }
  for (int i = tid; i < kH; i += kThreads) sb1[i] = b1[i];
  for (int i = tid; i < kD; i += kThreads) sb2[i] = b2[i];

  const int lane = tid & 31;
  const int wave = tid >> 5;
  const int mt   = wave >> 2;         // 0..1 : 16-row M tile of this CTA's 32 rows
  const int q    = wave & 3;          // 0..3 : 16-col N tile of the 64-wide state
  const int r15  = lane & 15;
  const int rhi  = (lane >> 4) << 3;  // 0 or 8 (C-layout row offset)
  const int row0 = mt * 16;
  const int nsub = nsubp[0];

  // ODE state, register-resident in native WMMA C-layout:
  // element r <-> (m = row0 + r + rhi, d = q*16 + r15)
  float yr[8];
  {
    const float* ysrc = y0 + (size_t)wg * kBM * kD;
    #pragma unroll
    for (int r = 0; r < 8; ++r)
      yr[r] = ysrc[(row0 + r + rhi) * kD + q * 16 + r15];
  }
  // save t = 0 (streaming / non-temporal: 134MB trajectory should not occupy L2)
  #pragma unroll
  for (int r = 0; r < 8; ++r) {
    size_t b = (size_t)wg * kBM + row0 + r + rhi;
    __builtin_nontemporal_store(yr[r], &out[(b * kT + 0) * kD + q * 16 + r15]);
  }
  __syncthreads();  // weights visible

  // Hoist ALL loop-invariant operands into registers: the B-side of both GEMMs
  // and both biases never change across the 1016-substep integration.
  v16h w1f[4][2];                     // GEMM1 B-frags: [n-tile][k-step]
  #pragma unroll
  for (int nt = 0; nt < 4; ++nt) {
    w1f[nt][0] = load_fragB(sW1t, LDW1, (q * 4 + nt) * 16, 0,  lane);
    w1f[nt][1] = load_fragB(sW1t, LDW1, (q * 4 + nt) * 16, 32, lane);
  }
  v16h w2f[8];                        // GEMM2 B-frags: [k-step]
  #pragma unroll
  for (int ks = 0; ks < 8; ++ks)
    w2f[ks] = load_fragB(sW2t, LDW2, q * 16, ks * 32, lane);
  float bb1v[4];
  #pragma unroll
  for (int nt = 0; nt < 4; ++nt) bb1v[nt] = sb1[(q * 4 + nt) * 16 + r15];
  const float bb2 = sb2[q * 16 + r15];

  for (int t = 0; t < kT - 1; ++t) {
    const float h = (ts[t + 1] - ts[t]) / (float)nsub;
    for (int sub = 0; sub < nsub; ++sub) {
      v8f kk[6];                      // six stage derivatives, register-resident
      #pragma unroll
      for (int st = 0; st < 6; ++st) {
        // ---- stage input S = y + h * sum_j a[st][j] * k_j  -> sA (f16) ----
        #pragma unroll
        for (int r = 0; r < 8; ++r) {
          float acc = 0.0f;
          if (st == 1) acc = cA21 * kk[0][r];
          if (st == 2) acc = cA31 * kk[0][r] + cA32 * kk[1][r];
          if (st == 3) acc = cA41 * kk[0][r] + cA42 * kk[1][r] + cA43 * kk[2][r];
          if (st == 4) acc = cA51 * kk[0][r] + cA52 * kk[1][r] + cA53 * kk[2][r] + cA54 * kk[3][r];
          if (st == 5) acc = cA61 * kk[0][r] + cA62 * kk[1][r] + cA63 * kk[2][r] + cA64 * kk[3][r] + cA65 * kk[4][r];
          sA[(row0 + r + rhi) * LDA + q * 16 + r15] = (_Float16)(yr[r] + h * acc);
        }
        __syncthreads();

        // ---- GEMM1: hidden = tanh(S @ W1 + b1) -> sAh (f16) ----
        v16h a0 = load_fragA(sA, LDA, row0, 0,  lane);
        v16h a1 = load_fragA(sA, LDA, row0, 32, lane);
        #pragma unroll
        for (int nt = 0; nt < 4; ++nt) {
          const int n = q * 4 + nt;                // H-tile 0..15
          const float bb = bb1v[nt];
          v8f c = {bb, bb, bb, bb, bb, bb, bb, bb};
          c = wmma_f16(a0, w1f[nt][0], c);
          c = wmma_f16(a1, w1f[nt][1], c);
          // packed-f16 tanh; pairs (r, r+1) -> one low store + one high store
          #pragma unroll
          for (int rp = 0; rp < 4; ++rp) {
            __half2 hv = tanh2(c[2 * rp], c[2 * rp + 1]);
            __half* p0 = (__half*)&sAh[(row0 + 2 * rp     + rhi) * LDH + n * 16 + r15];
            __half* p1 = (__half*)&sAh[(row0 + 2 * rp + 1 + rhi) * LDH + n * 16 + r15];
            *p0 = __low2half(hv);     // ds_store_b16
            *p1 = __high2half(hv);    // ds_store_b16_d16_hi
          }
        }
        __syncthreads();

        // ---- GEMM2: k_st = hidden @ W2 + b2 (stays in registers) ----
        v8f c = {bb2, bb2, bb2, bb2, bb2, bb2, bb2, bb2};
        #pragma unroll
        for (int ks = 0; ks < 8; ++ks) {
          v16h a = load_fragA(sAh, LDH, row0, ks * 32, lane);
          c = wmma_f16(a, w2f[ks], c);
        }
        kk[st] = c;
      }
      // ---- 5th-order update ----
      #pragma unroll
      for (int r = 0; r < 8; ++r)
        yr[r] += h * (cB1 * kk[0][r] + cB2 * kk[1][r] + cB3 * kk[2][r] +
                      cB4 * kk[3][r] + cB5 * kk[4][r] + cB6 * kk[5][r]);
    }
    // save at t+1
    #pragma unroll
    for (int r = 0; r < 8; ++r) {
      size_t b = (size_t)wg * kBM + row0 + r + rhi;
      __builtin_nontemporal_store(yr[r], &out[(b * kT + (t + 1)) * kD + q * 16 + r15]);
    }
  }
}

extern "C" void kernel_launch(void* const* d_in, const int* in_sizes, int n_in,
                              void* d_out, int out_size, void* d_ws, size_t ws_size,
                              hipStream_t stream) {
  const float* y0 = (const float*)d_in[0];
  const float* ts = (const float*)d_in[1];
  const float* W1 = (const float*)d_in[2];
  const float* b1 = (const float*)d_in[3];
  const float* W2 = (const float*)d_in[4];
  const float* b2 = (const float*)d_in[5];
  const int* nsub = (const int*)d_in[6];
  float* out = (float*)d_out;

  dim3 grid(kBatch / kBM);   // 128 persistent CTAs
  dim3 block(kThreads);      // 8 wave32
  node_tsit5_kernel<<<grid, block, kSmemBytes, stream>>>(y0, ts, W1, b1, W2, b2, nsub, out);
}
